// ModelLinkPred_weight_38173669327419
// MI455X (gfx1250) — matrix operand order repaired and checked by reading, not compiled, not run on hardware
//
#include <hip/hip_runtime.h>
#include <hip/hip_bf16.h>

typedef __attribute__((ext_vector_type(16))) __bf16 v16bf;
typedef __attribute__((ext_vector_type(8)))  __bf16 v8bf;
typedef __attribute__((ext_vector_type(8)))  float  v8f;

#define HDIM 128
#define NPART 32
#define BN_EPS 1e-5f

__device__ __forceinline__ void atomAddF(float* p, float v) { unsafeAtomicAdd(p, v); }

// ---------------- degree / norm prep ----------------
__global__ void k_init_deg(float* deg, int n) {
    int i = blockIdx.x * blockDim.x + threadIdx.x;
    if (i < n) deg[i] = 1.0f;               // self-loop weight
}
__global__ void k_deg_accum(const int* __restrict__ dst, const float* __restrict__ ew,
                            float* __restrict__ deg, int e) {
    int i = blockIdx.x * blockDim.x + threadIdx.x;
    if (i < e) atomAddF(&deg[dst[i]], ew[i]);
}
__global__ void k_dinv(float* deg, int n) {
    int i = blockIdx.x * blockDim.x + threadIdx.x;
    if (i < n) deg[i] = rsqrtf(deg[i]);     // in place: deg -> dinv
}
__global__ void k_norm(const int* __restrict__ src, const int* __restrict__ dst,
                       const float* __restrict__ ew, const float* __restrict__ dinv,
                       float* __restrict__ norm, int e) {
    int i = blockIdx.x * blockDim.x + threadIdx.x;
    if (i < e) norm[i] = dinv[src[i]] * ew[i] * dinv[dst[i]];
}

// ---------------- conversions ----------------
__global__ void k_f2bf(const float* __restrict__ in, __bf16* __restrict__ out, int total) {
    int i = blockIdx.x * blockDim.x + threadIdx.x;
    if (i < total) out[i] = (__bf16)in[i];
}
// W [128(k) x 128(j)] row-major f32 -> Wt [j][k] bf16 (column-major weight for B frags)
__global__ void k_wT_bf(const float* __restrict__ W, __bf16* __restrict__ Wt) {
    int i = blockIdx.x * blockDim.x + threadIdx.x;   // 16384
    int j = i & 127, k = i >> 7;
    Wt[j * HDIM + k] = (__bf16)W[k * HDIM + j];
}

// ---------------- WMMA GEMM: C[N,128] = A(bf16)[N,128] @ W + bias ----------------
// 1 wave -> 16x16 tile; 8 waves/block -> 16 rows x 128 cols; grid = ceil(N/16)
// A frag (16x32 bf16): lane half hi: K in {kbA..kbA+7} U {16+kbA..16+kbA+7}, kbA = 8*hi
// B frag (32x16 bf16): lane half hi: K in {kbA..kbA+7} U {16+kbA..16+kbA+7}, same as A
// D frag (8 VGPRs f32): m = r + 8*hi, n = lane&15
__global__ void k_gemm_bf16(const __bf16* __restrict__ A, const __bf16* __restrict__ Bt,
                            const float* __restrict__ bias, float* __restrict__ C, int nrows) {
    const int wave = threadIdx.x >> 5;
    const int lane = threadIdx.x & 31;
    const int rowBase = blockIdx.x * 16;
    if (rowBase >= nrows) return;
    const int colBase = wave * 16;
    const int m  = lane & 15;
    const int hi = lane >> 4;            // lane half
    const int kbA = hi * 8;
    const int kbB = hi * 16;
    int rA = rowBase + m; if (rA >= nrows) rA = nrows - 1;   // tail-safe load
    const __bf16* arow = A  + (long long)rA * HDIM;
    const __bf16* bcol = Bt + (colBase + m) * HDIM;          // lane&15 is also the B column
    v8f c = {};
#pragma unroll
    for (int kt = 0; kt < 4; ++kt) {
        const int k0 = kt * 32;
        const v8bf a_lo = *(const v8bf*)(arow + k0 + kbA);
        const v8bf a_hi = *(const v8bf*)(arow + k0 + 16 + kbA);
        const v8bf b_lo = *(const v8bf*)(bcol + k0 + kbA);
        const v8bf b_hi = *(const v8bf*)(bcol + k0 + 16 + kbA);
        v16bf a, b;
#pragma unroll
        for (int j = 0; j < 8; ++j) {
            a[j] = a_lo[j]; a[j + 8] = a_hi[j];
            b[j] = b_lo[j]; b[j + 8] = b_hi[j];
        }
        c = __builtin_amdgcn_wmma_f32_16x16x32_bf16(false, a, false, b, (short)0, c, false, false);
    }
    const int n = lane & 15;
    const int mOff = hi * 8;
    const float bv = bias[colBase + n];
    float* cptr = C + (long long)(rowBase + mOff) * HDIM + colBase + n;
    if (rowBase + 16 <= nrows) {
        // full tile: uniform branch, straight-line 8-store epilogue
#pragma unroll
        for (int r = 0; r < 8; ++r) cptr[r * HDIM] = c[r] + bv;
    } else {
#pragma unroll
        for (int r = 0; r < 8; ++r)
            if (rowBase + mOff + r < nrows) cptr[r * HDIM] = c[r] + bv;
    }
}

// ---------------- self-loop init + edge scatter ----------------
__global__ void k_selfloop(const float* __restrict__ hw, const float* __restrict__ dinv,
                           float* __restrict__ agg, int total) {
    int i = blockIdx.x * blockDim.x + threadIdx.x;
    if (i >= total) return;
    const float d = dinv[i >> 7];
    agg[i] = hw[i] * d * d;
}
// 32 lanes per edge, each lane owns 4 features; hw/agg stay L2-resident (25.6 MB << 192 MB)
__global__ void k_scatter(const int* __restrict__ src, const int* __restrict__ dst,
                          const float* __restrict__ norm, const float* __restrict__ hw,
                          float* __restrict__ agg, int nedges) {
    const int gid = blockIdx.x * blockDim.x + threadIdx.x;
    const int e = gid >> 5, g = gid & 31;
    if (e >= nedges) return;
    const float w = norm[e];
    const int s = src[e], d = dst[e];
    const float4 v = *(const float4*)(hw + (long long)s * HDIM + g * 4);
    float* o = agg + (long long)d * HDIM + g * 4;
    atomAddF(o + 0, v.x * w);
    atomAddF(o + 1, v.y * w);
    atomAddF(o + 2, v.z * w);
    atomAddF(o + 3, v.w * w);
}

// ---------------- BatchNorm (biased var) + ReLU ----------------
__global__ void k_bn_zero(float* sum, float* sumsq) {
    int i = threadIdx.x;
    if (i < HDIM) { sum[i] = 0.f; sumsq[i] = 0.f; }
}
__global__ void k_bn_partial(const float* __restrict__ agg, float* __restrict__ sum,
                             float* __restrict__ sumsq, int nrows) {
    __shared__ float sS[256], sQ[256];
    const int t = threadIdx.x;
    const int c = t & 127;
    const int r0 = blockIdx.x * 256 + (t >> 7);
    float s = 0.f, q = 0.f;
#pragma unroll 4
    for (int j = 0; j < 128; ++j) {
        const int r = r0 + j * 2;
        if (r < nrows) {
            // stream ahead: predictable pattern, keep the L2->WGP pipe busy
            if (r + 64 < nrows) __builtin_prefetch(&agg[(long long)(r + 64) * HDIM + c], 0, 1);
            const float v = agg[(long long)r * HDIM + c];
            s += v; q += v * v;
        }
    }
    sS[t] = s; sQ[t] = q;
    __syncthreads();
    if (t < 128) {
        atomAddF(&sum[c],   sS[t] + sS[t + 128]);
        atomAddF(&sumsq[c], sQ[t] + sQ[t + 128]);
    }
}
__global__ void k_bn_finalize(const float* sum, const float* sumsq,
                              float* mean, float* rstd, int nrows) {
    int c = threadIdx.x;
    if (c >= HDIM) return;
    const float invN = 1.0f / (float)nrows;
    const float m = sum[c] * invN;
    const float v = sumsq[c] * invN - m * m;
    mean[c] = m;
    rstd[c] = rsqrtf(v + BN_EPS);
}
__global__ void k_bn_apply(const float* __restrict__ agg, const float* __restrict__ mean,
                           const float* __restrict__ rstd, __bf16* __restrict__ hb,
                           float* __restrict__ hf, int writeF, int total) {
    int i = blockIdx.x * blockDim.x + threadIdx.x;
    if (i >= total) return;
    const int c = i & 127;
    float v = (agg[i] - mean[c]) * rstd[c];
    v = fmaxf(v, 0.f);
    hb[i] = (__bf16)v;
    if (writeF) hf[i] = v;
}

// ---------------- MLP pieces ----------------
// vvec[j] = x_curr @ lin1_w[128:256, j];  csum[j] = sum_k lin1_w[256:512, j]
__global__ void k_mlp_prep(const float* __restrict__ hF, const int* __restrict__ currId,
                           const float* __restrict__ w1, float* __restrict__ vvec,
                           float* __restrict__ csum) {
    const int j = threadIdx.x;
    if (j >= HDIM) return;
    const float* xc = hF + (long long)currId[0] * HDIM;
    float v = 0.f;
    for (int k = 0; k < 128; ++k) v += xc[k] * w1[(128 + k) * HDIM + j];
    float cs = 0.f;
    for (int k = 256; k < 512; ++k) cs += w1[k * HDIM + j];
    vvec[j] = v;
    csum[j] = cs;
}
__global__ void k_zero_out(float* out) {
    int i = threadIdx.x;
    if (i < NPART) out[i] = 0.f;
}
// one wave per node: score = b2 + sum_j relu(t[n,j]+v[j]+ecn[n]*cs[j]) * w2[j]
// then out[p] += score * partitions[n,p]  (lane p handles partition p)
__global__ void k_score(const float* __restrict__ t, const float* __restrict__ vvec,
                        const float* __restrict__ csum, const float* __restrict__ ecn,
                        const float* __restrict__ w2, const float* __restrict__ b2,
                        const float* __restrict__ parts, float* __restrict__ out, int nrows) {
    const int wid = (blockIdx.x * blockDim.x + threadIdx.x) >> 5;
    const int lane = threadIdx.x & 31;
    if (wid >= nrows) return;
    const float e = ecn[wid];
    float acc = 0.f;
#pragma unroll
    for (int q = 0; q < 4; ++q) {
        const int j = lane + q * 32;
        const float x = t[(long long)wid * HDIM + j] + vvec[j] + e * csum[j];
        acc += fmaxf(x, 0.f) * w2[j];
    }
    for (int off = 16; off; off >>= 1) acc += __shfl_xor(acc, off, 32);
    const float score = acc + b2[0];
    atomAddF(&out[lane], score * parts[(long long)wid * NPART + lane]);
}

// ---------------- host launch ----------------
static inline size_t alignUp(size_t x) { return (x + 255) & ~(size_t)255; }

extern "C" void kernel_launch(void* const* d_in, const int* in_sizes, int n_in,
                              void* d_out, int out_size, void* d_ws, size_t ws_size,
                              hipStream_t stream) {
    const float* x      = (const float*)d_in[0];
    const int*   ei     = (const int*)d_in[1];
    const float* ew     = (const float*)d_in[2];
    const float* parts  = (const float*)d_in[3];
    const float* ecn    = (const float*)d_in[5];
    const float* conv_w = (const float*)d_in[6];
    const float* conv_b = (const float*)d_in[7];
    const float* lin1_w = (const float*)d_in[8];
    const float* lin1_b = (const float*)d_in[9];
    const float* lin2_w = (const float*)d_in[10];
    const float* lin2_b = (const float*)d_in[11];
    const int*   currId = (const int*)d_in[12];

    const int N = in_sizes[0] / HDIM;
    const int E = in_sizes[2];
    const int* src = ei;
    const int* dst = ei + E;

    float* outP = (float*)d_out;            // [32] partition scores
    float* outH = (float*)d_out + NPART;    // [N,128] final h (f32)

    char* ws = (char*)d_ws;
    size_t o = 0;
    float*  DINV = (float*)(ws + o);  o = alignUp(o + (size_t)N * 4);
    float*  NORM = (float*)(ws + o);  o = alignUp(o + (size_t)E * 4);
    __bf16* HB   = (__bf16*)(ws + o); o = alignUp(o + (size_t)N * HDIM * 2);
    float*  HW   = (float*)(ws + o);  o = alignUp(o + (size_t)N * HDIM * 4);
    float*  AGG  = (float*)(ws + o);  o = alignUp(o + (size_t)N * HDIM * 4);
    __bf16* WT   = (__bf16*)(ws + o); o = alignUp(o + (size_t)HDIM * HDIM * 2);
    float*  BSUM = (float*)(ws + o);  o = alignUp(o + 4 * HDIM * 4);
    float*  BSQ  = BSUM + HDIM;
    float*  BMEAN= BSUM + 2 * HDIM;
    float*  BRSTD= BSUM + 3 * HDIM;
    float*  VV   = (float*)(ws + o);  o = alignUp(o + 2 * HDIM * 4);
    float*  CS   = VV + HDIM;
    (void)ws_size; (void)n_in; (void)out_size;

    const int TB = 256;
    const int gN   = (N + TB - 1) / TB;
    const int gE   = (E + TB - 1) / TB;
    const int gNH  = (N * HDIM + TB - 1) / TB;
    const int gE32 = (E * 32 + TB - 1) / TB;       // 32 lanes per edge (4 floats each)
    const int gT16 = (N + 15) / 16;                // GEMM row tiles
    const int gBN  = (N + 255) / 256;              // BN partial blocks
    const int gWv  = (N * 32 + TB - 1) / TB;       // wave-per-row score kernel

    k_zero_out<<<1, 32, 0, stream>>>(outP);

    // degree / normalization
    k_init_deg<<<gN, TB, 0, stream>>>(DINV, N);
    k_deg_accum<<<gE, TB, 0, stream>>>(dst, ew, DINV, E);
    k_dinv<<<gN, TB, 0, stream>>>(DINV, N);
    k_norm<<<gE, TB, 0, stream>>>(src, dst, ew, DINV, NORM, E);

    // h0 = x in bf16
    k_f2bf<<<gNH, TB, 0, stream>>>(x, HB, N * HDIM);

    for (int l = 0; l < 3; ++l) {
        k_wT_bf<<<(HDIM * HDIM) / TB, TB, 0, stream>>>(conv_w + (size_t)l * HDIM * HDIM, WT);
        k_gemm_bf16<<<gT16, TB, 0, stream>>>(HB, WT, conv_b + (size_t)l * HDIM, HW, N);
        k_selfloop<<<gNH, TB, 0, stream>>>(HW, DINV, AGG, N * HDIM);
        k_scatter<<<gE32, TB, 0, stream>>>(src, dst, NORM, HW, AGG, E);
        k_bn_zero<<<1, HDIM, 0, stream>>>(BSUM, BSQ);
        k_bn_partial<<<gBN, TB, 0, stream>>>(AGG, BSUM, BSQ, N);
        k_bn_finalize<<<1, HDIM, 0, stream>>>(BSUM, BSQ, BMEAN, BRSTD, N);
        k_bn_apply<<<gNH, TB, 0, stream>>>(AGG, BMEAN, BRSTD, HB, outH, (l == 2) ? 1 : 0, N * HDIM);
    }

    // MLP: t = h @ lin1_w[0:128] + lin1_b ; broadcast cols collapsed to vvec/csum
    k_mlp_prep<<<1, HDIM, 0, stream>>>(outH, currId, lin1_w, VV, CS);
    k_wT_bf<<<(HDIM * HDIM) / TB, TB, 0, stream>>>(lin1_w, WT);   // rows 0..127 of [512,128]
    k_gemm_bf16<<<gT16, TB, 0, stream>>>(HB, WT, lin1_b, HW, N);
    k_score<<<gWv, TB, 0, stream>>>(HW, VV, CS, ecn, lin2_w, lin2_b, parts, outP, N);
}
